// OneScaleMultiStepPredictor_75823352644381
// MI455X (gfx1250) — matrix operand order repaired and checked by reading, not compiled, not run on hardware
//
#include <hip/hip_runtime.h>
#include <hip/hip_bf16.h>

// ---------------------------------------------------------------------------
// MI455X (gfx1250, wave32) OneScaleMultiStepPredictor.
// - All GEMMs on v_wmma_f32_16x16x32_f16, f32 accumulation.
// - All intermediate activations stored f16 (memory-bound workload: halves
//   the dominant HBM traffic; WMMA consumes f16 anyway).
// - Tile staging uses gfx1250 async global->LDS copies
//   (global_load_async_to_lds_b128 / s_wait_asynccnt).
// - Sparse conv skips taps where the whole 16-row tile has no neighbor.
// ---------------------------------------------------------------------------

typedef __attribute__((ext_vector_type(16))) _Float16 v16h;
typedef __attribute__((ext_vector_type(8)))  _Float16 v8h;
typedef __attribute__((ext_vector_type(8)))  float    v8f;

// ---- WMMA helpers ----------------------------------------------------------

__device__ __forceinline__ v8f wmma32(v16h a, v16h b, v8f c) {
    return __builtin_amdgcn_wmma_f32_16x16x32_f16(false, a, false, b,
                                                  (short)0, c, false, false);
}

// A fragment (16x32 f16, ISA 7.12.2 layout).
__device__ __forceinline__ v16h load_a_frag(const _Float16* Alds, int ldA,
                                            int lane, int kk) {
    int half = lane >> 4;
    int row  = lane & 15;
    const _Float16* base = Alds + row * ldA + kk + half * 8;
    v8h lo = *(const v8h*)(base);
    v8h hi = *(const v8h*)(base + 16);
    v16h a;
#pragma unroll
    for (int i = 0; i < 8; ++i) { a[i] = lo[i]; a[8 + i] = hi[i]; }
    return a;
}

// B fragment: pre-packed, one aligned 32B load per lane.
__device__ __forceinline__ v16h load_b_frag(const _Float16* pWblk, int lane) {
    return *(const v16h*)(pWblk + lane * 16);
}

// ---- gfx1250 async global->LDS (ASYNCcnt-tracked) --------------------------

__device__ __forceinline__ void async_copy16(const _Float16* gsrc,
                                             _Float16* ldst) {
    unsigned lds = (unsigned)(unsigned long long)(uintptr_t)ldst;
    unsigned long long ga = (unsigned long long)(uintptr_t)gsrc;
    asm volatile("global_load_async_to_lds_b128 %0, %1, off"
                 :: "v"(lds), "v"(ga) : "memory");
}
__device__ __forceinline__ void wait_async0() {
    asm volatile("s_wait_asynccnt 0" ::: "memory");
}

// ---- weight packing --------------------------------------------------------
__global__ void k_pack(const float* __restrict__ W, _Float16* __restrict__ dst,
                       int K, int Ncols, int chunks, int NT, int nmat) {
    long per   = (long)chunks * NT * 512;
    long total = per * nmat;
    for (long e = (long)blockIdx.x * blockDim.x + threadIdx.x; e < total;
         e += (long)gridDim.x * blockDim.x) {
        int  b = (int)(e / per);
        long r = e % per;
        int chunk = (int)(r / ((long)NT * 512)); r %= (long)NT * 512;
        int ntile = (int)(r / 512);              r %= 512;
        int lane  = (int)(r / 16);
        int i     = (int)(r % 16);
        int sk = chunk * 32 + (lane >> 4) * 8 + (i & 7) + (i >> 3) * 16;
        int sn = ntile * 16 + (lane & 15);
        float v = (sk < K && sn < Ncols)
                      ? W[(long)b * K * Ncols + (long)sk * Ncols + sn]
                      : 0.0f;
        dst[e] = (_Float16)v;
    }
}

__global__ void k_zero(float* p) {
    if (blockIdx.x == 0 && threadIdx.x == 0) p[0] = 0.0f;
}

// ---- dec: h16 = PReLU(concat(rec_F, bins1_F) @ W_dec + b_dec) --------------
__global__ void k_dec(const float* __restrict__ recF,
                      const float* __restrict__ bins1,
                      const _Float16* __restrict__ pW,
                      const float* __restrict__ bias,
                      const float* __restrict__ alpha,
                      _Float16* __restrict__ out16, int N) {
    __shared__ __align__(32) _Float16 Alds[16 * 96];
    int row0 = blockIdx.x * 16;
    int tid  = threadIdx.x;
    for (int e = tid; e < 16 * 96; e += 128) {
        int r = e / 96, c = e % 96;
        int gr = row0 + r;
        float v = 0.0f;
        if (gr < N) {
            if (c < 64)      v = recF[(long)gr * 64 + c];
            else if (c < 72) v = bins1[(long)gr * 8 + (c - 64)];
        }
        Alds[r * 96 + c] = (_Float16)v;
    }
    __syncthreads();
    int wave = tid >> 5, lane = tid & 31;
    v8f acc = {};
#pragma unroll
    for (int chunk = 0; chunk < 3; ++chunk) {   // K = 72 padded to 96
        v16h a = load_a_frag(Alds, 96, lane, chunk * 32);
        v16h b = load_b_frag(pW + (chunk * 4 + wave) * 512, lane);
        acc = wmma32(a, b, acc);
    }
    float al   = alpha[0];
    int   col  = wave * 16 + (lane & 15);
    int   mb   = (lane >> 4) << 3;
    float bcol = bias[col];
    if (row0 + 16 <= N) {                        // full-tile fast path
#pragma unroll
        for (int r = 0; r < 8; ++r) {
            float v = acc[r] + bcol;
            v = (v >= 0.0f) ? v : al * v;
            out16[(long)(row0 + mb + r) * 64 + col] = (_Float16)v;
        }
    } else {
#pragma unroll
        for (int r = 0; r < 8; ++r) {
            int gr = row0 + mb + r;
            if (gr < N) {
                float v = acc[r] + bcol;
                v = (v >= 0.0f) ? v : al * v;
                out16[(long)gr * 64 + col] = (_Float16)v;
            }
        }
    }
}

// ---- sparse 3x3x3 conv over f16 features -----------------------------------
// out16[i] = act(b + sum_k W[k]^T x16[nbr[i,k]] (+res16));  out32 optional
__global__ void k_sconv(const _Float16* __restrict__ x,
                        const int* __restrict__ nbr,
                        const _Float16* __restrict__ pW,   // [27][2][4][512]
                        const float* __restrict__ bias,
                        const float* __restrict__ alpha,
                        const _Float16* __restrict__ res16,  // optional
                        _Float16* __restrict__ out16,
                        float* __restrict__ out32,           // optional
                        int rows) {
    __shared__ int sidx[2][16];
    __shared__ __align__(32) _Float16 Alds[2][16 * 64];
    int row0 = blockIdx.x * 16;
    int tid  = threadIdx.x;
    int wave = tid >> 5, lane = tid & 31;
    v8f acc = {};
    for (int k = 0; k < 27; ++k) {
        int p = k & 1;
        if (tid < 16) {
            int gr = row0 + tid;
            sidx[p][tid] = (gr < rows) ? nbr[(long)gr * 27 + k] : -1;
        }
        if (tid == 0 && k + 1 < 27)
            __builtin_prefetch(pW + (long)(k + 1) * 4096, 0, 0);
        __syncthreads();
        bool any = __ballot(sidx[p][lane & 15] >= 0) != 0;  // uniform
        if (any) {
            // stage 16x64 f16 tile: 128 threads x 16B async copies
            int r = tid >> 3, seg = tid & 7;
            int idx = sidx[p][r];
            _Float16* ldst = &Alds[p][r * 64 + seg * 8];
            if (idx >= 0) {
                async_copy16(x + (long)idx * 64 + seg * 8, ldst);
            } else {
                v8h z = {};
                *(v8h*)ldst = z;
            }
            wait_async0();
            __syncthreads();
            const _Float16* pWk = pW + (long)k * 4096;
#pragma unroll
            for (int chunk = 0; chunk < 2; ++chunk) {
                v16h a = load_a_frag(Alds[p], 64, lane, chunk * 32);
                v16h b = load_b_frag(pWk + (chunk * 4 + wave) * 512, lane);
                acc = wmma32(a, b, acc);
            }
        }
    }
    float al   = alpha[0];
    int   col  = wave * 16 + (lane & 15);
    int   mb   = (lane >> 4) << 3;
    float bcol = bias[col];
    if (row0 + 16 <= rows) {                     // full-tile fast path
#pragma unroll
        for (int r = 0; r < 8; ++r) {
            long o = (long)(row0 + mb + r) * 64 + col;
            float v = acc[r] + bcol;
            if (res16) v += (float)res16[o];
            v = (v >= 0.0f) ? v : al * v;
            out16[o] = (_Float16)v;
            if (out32) out32[o] = v;
        }
    } else {
#pragma unroll
        for (int r = 0; r < 8; ++r) {
            int gr = row0 + mb + r;
            if (gr < rows) {
                long o = (long)gr * 64 + col;
                float v = acc[r] + bcol;
                if (res16) v += (float)res16[o];
                v = (v >= 0.0f) ? v : al * v;
                out16[o] = (_Float16)v;
                if (out32) out32[o] = v;
            }
        }
    }
}

// ---- 64 -> 512 linear (pred[0] head), f16 out ------------------------------
__global__ void k_lin8(const _Float16* __restrict__ x,
                       const _Float16* __restrict__ pW,    // [2][32][512]
                       const float* __restrict__ bias,
                       _Float16* __restrict__ out, int N) {
    __shared__ __align__(32) _Float16 Alds[16 * 64];
    int row0 = blockIdx.x * 16, tid = threadIdx.x;
    {
        int r = tid >> 3, seg = tid & 7;
        int gr = row0 + r;
        _Float16* ldst = &Alds[r * 64 + seg * 8];
        if (gr < N) {
            async_copy16(x + (long)gr * 64 + seg * 8, ldst);
        } else {
            v8h z = {};
            *(v8h*)ldst = z;
        }
    }
    wait_async0();
    __syncthreads();
    int wave = tid >> 5, lane = tid & 31;
    bool full = (row0 + 16 <= N);
    int  mb   = (lane >> 4) << 3;
#pragma unroll
    for (int j = 0; j < 8; ++j) {               // 4 waves x 8 = 32 col-tiles
        int ntile = wave * 8 + j;
        v8f acc = {};
#pragma unroll
        for (int chunk = 0; chunk < 2; ++chunk) {
            v16h a = load_a_frag(Alds, 64, lane, chunk * 32);
            v16h b = load_b_frag(pW + (chunk * 32 + ntile) * 512, lane);
            acc = wmma32(a, b, acc);
        }
        int col = ntile * 16 + (lane & 15);
        float bcol = bias[col];
        if (full) {
#pragma unroll
            for (int r = 0; r < 8; ++r)
                out[(long)(row0 + mb + r) * 512 + col] = (_Float16)(acc[r] + bcol);
        } else {
#pragma unroll
            for (int r = 0; r < 8; ++r) {
                int gr = row0 + mb + r;
                if (gr < N) out[(long)gr * 512 + col] = (_Float16)(acc[r] + bcol);
            }
        }
    }
}

// ---- bool-mask gather: g[m] = l8[child_pidx[m], child_bit[m], :] (16B/thr) -
__global__ void k_gather(const _Float16* __restrict__ l8,
                         const int* __restrict__ pidx,
                         const int* __restrict__ bit,
                         _Float16* __restrict__ g, int M) {
    long e = (long)blockIdx.x * 256 + threadIdx.x;   // one 16B chunk each
    if (e < (long)M * 8) {
        int m = (int)(e >> 3), seg = (int)(e & 7);
        const v8h* src =
            (const v8h*)(l8 + (long)pidx[m] * 512 + (long)bit[m] * 64 + seg * 8);
        *(v8h*)(g + (long)m * 64 + seg * 8) = *src;
    }
}

// ---- logits (64->255) + log-softmax CE + scaled atomic loss ----------------
__global__ void k_logits_ce(const _Float16* __restrict__ q,
                            const _Float16* __restrict__ pW,  // [2][16][512]
                            const float* __restrict__ bias,   // [255]
                            const int* __restrict__ bins0,    // [M][8]
                            const int* __restrict__ octk,     // [8]
                            const int* __restrict__ refb,     // [M]
                            const float* __restrict__ pnum,   // [BATCH]
                            int M, float scale,               // LOG2E/BATCH
                            float* __restrict__ loss) {
    __shared__ __align__(32) _Float16 Alds[16 * 64];
    __shared__ float Llds[16 * 256];
    int row0 = blockIdx.x * 16, tid = threadIdx.x;
    {
        int r = tid >> 3, seg = tid & 7;
        int gr = row0 + r;
        _Float16* ldst = &Alds[r * 64 + seg * 8];
        if (gr < M) {
            async_copy16(q + (long)gr * 64 + seg * 8, ldst);
        } else {
            v8h z = {};
            *(v8h*)ldst = z;
        }
    }
    wait_async0();
    __syncthreads();
    int wave = tid >> 5, lane = tid & 31;
#pragma unroll
    for (int j = 0; j < 4; ++j) {               // 4 waves x 4 = 16 col-tiles
        int ntile = wave * 4 + j;
        v8f acc = {};
#pragma unroll
        for (int chunk = 0; chunk < 2; ++chunk) {
            v16h a = load_a_frag(Alds, 64, lane, chunk * 32);
            v16h b = load_b_frag(pW + (chunk * 16 + ntile) * 512, lane);
            acc = wmma32(a, b, acc);
        }
        int col = ntile * 16 + (lane & 15);
        float bc = (col < 255) ? bias[col] : 0.0f;
#pragma unroll
        for (int r = 0; r < 8; ++r) {
            int m = ((lane >> 4) << 3) + r;
            Llds[m * 256 + col] = acc[r] + bc;
        }
    }
    __syncthreads();
    if (tid < 16) {
        int gr = row0 + tid;
        if (gr < M) {
            const float* L = &Llds[tid * 256];
            float mx = L[0];
            for (int c = 1; c < 255; ++c) mx = fmaxf(mx, L[c]);
            float s = 0.0f;
            for (int c = 0; c < 255; ++c) s += __expf(L[c] - mx);
            int oct = 0;
#pragma unroll
            for (int jj = 0; jj < 8; ++jj)
                oct += bins0[(long)gr * 8 + jj] << octk[jj];
            oct -= 1;
            float ce = -(L[oct] - mx - __logf(s));
            atomicAdd(loss, (ce / pnum[refb[gr]]) * scale);
        }
    }
}

// ---------------------------------------------------------------------------

extern "C" void kernel_launch(void* const* d_in, const int* in_sizes, int n_in,
                              void* d_out, int out_size, void* d_ws, size_t ws_size,
                              hipStream_t stream) {
    const float* rec_F   = (const float*)d_in[0];
    const float* bins1_F = (const float*)d_in[1];
    const float* pnum    = (const float*)d_in[2];
    const float* W_dec   = (const float*)d_in[3];
    const float* b_dec   = (const float*)d_in[4];
    const float* a_dec   = (const float*)d_in[5];
    const float* Wb1     = (const float*)d_in[6];
    const float* bb1     = (const float*)d_in[7];
    const float* ab1     = (const float*)d_in[8];
    const float* Wb2     = (const float*)d_in[9];
    const float* bb2     = (const float*)d_in[10];
    const float* ab2     = (const float*)d_in[11];
    const float* Wp0c    = (const float*)d_in[12];
    const float* bp0c    = (const float*)d_in[13];
    const float* ap0     = (const float*)d_in[14];
    const float* Wp0l    = (const float*)d_in[15];
    const float* bp0l    = (const float*)d_in[16];
    const float* Wp1c    = (const float*)d_in[17];
    const float* bp1c    = (const float*)d_in[18];
    const float* ap1     = (const float*)d_in[19];
    const float* Wp1l    = (const float*)d_in[20];
    const float* bp1l    = (const float*)d_in[21];
    const int*   nbr_rec    = (const int*)d_in[22];
    const int*   nbr_child  = (const int*)d_in[23];
    const int*   child_pidx = (const int*)d_in[24];
    const int*   child_bit  = (const int*)d_in[25];
    const int*   bins0      = (const int*)d_in[26];
    const int*   oct_kernel = (const int*)d_in[27];
    const int*   ref_batch  = (const int*)d_in[28];

    const int N     = in_sizes[0] / 64;
    const int M     = in_sizes[24];
    const int BATCH = in_sizes[2];

    float* y_out = (float*)d_out;                 // [N,64]
    float* loss  = (float*)d_out + (out_size - 1);

    char*  ws  = (char*)d_ws;
    size_t off = 0;
    auto carve = [&](size_t bytes) -> void* {
        off = (off + 255) & ~(size_t)255;
        void* p = ws + off;
        off += bytes;
        return p;
    };
    _Float16* pWdec = (_Float16*)carve((size_t)3 * 4 * 512 * 2);
    _Float16* pWb1  = (_Float16*)carve((size_t)27 * 2 * 4 * 512 * 2);
    _Float16* pWb2  = (_Float16*)carve((size_t)27 * 2 * 4 * 512 * 2);
    _Float16* pWp0c = (_Float16*)carve((size_t)27 * 2 * 4 * 512 * 2);
    _Float16* pWp1c = (_Float16*)carve((size_t)27 * 2 * 4 * 512 * 2);
    _Float16* pWp0l = (_Float16*)carve((size_t)2 * 32 * 512 * 2);
    _Float16* pWp1l = (_Float16*)carve((size_t)2 * 16 * 512 * 2);
    _Float16* h16  = (_Float16*)carve((size_t)N * 64 * 2);
    _Float16* t16  = (_Float16*)carve((size_t)N * 64 * 2);
    _Float16* y16  = (_Float16*)carve((size_t)N * 64 * 2);
    _Float16* p16  = (_Float16*)carve((size_t)N * 64 * 2);
    _Float16* l8   = (_Float16*)carve((size_t)N * 512 * 2);
    _Float16* g16  = (_Float16*)carve((size_t)M * 64 * 2);
    _Float16* q16  = (_Float16*)carve((size_t)M * 64 * 2);
    (void)ws_size;

    k_zero<<<1, 32, 0, stream>>>(loss);

    auto pack_grid = [](long total) { return dim3((unsigned)((total + 255) / 256)); };
    k_pack<<<pack_grid((long)3 * 4 * 512), 256, 0, stream>>>(W_dec, pWdec, 72, 64, 3, 4, 1);
    k_pack<<<pack_grid((long)27 * 2 * 4 * 512), 256, 0, stream>>>(Wb1, pWb1, 64, 64, 2, 4, 27);
    k_pack<<<pack_grid((long)27 * 2 * 4 * 512), 256, 0, stream>>>(Wb2, pWb2, 64, 64, 2, 4, 27);
    k_pack<<<pack_grid((long)27 * 2 * 4 * 512), 256, 0, stream>>>(Wp0c, pWp0c, 64, 64, 2, 4, 27);
    k_pack<<<pack_grid((long)27 * 2 * 4 * 512), 256, 0, stream>>>(Wp1c, pWp1c, 64, 64, 2, 4, 27);
    k_pack<<<pack_grid((long)2 * 32 * 512), 256, 0, stream>>>(Wp0l, pWp0l, 64, 512, 2, 32, 1);
    k_pack<<<pack_grid((long)2 * 16 * 512), 256, 0, stream>>>(Wp1l, pWp1l, 64, 255, 2, 16, 1);

    dim3 blkN((N + 15) / 16), blkM((M + 15) / 16), thr(128);

    // h = PReLU(concat(rec_F, bins1_F) @ W_dec + b_dec)
    k_dec<<<blkN, thr, 0, stream>>>(rec_F, bins1_F, pWdec, b_dec, a_dec, h16, N);
    // t = PReLU(sconv(h))
    k_sconv<<<blkN, thr, 0, stream>>>(h16, nbr_rec, pWb1, bb1, ab1,
                                      nullptr, t16, nullptr, N);
    // y = PReLU(sconv(t) + h)  -> f32 output + f16 copy
    k_sconv<<<blkN, thr, 0, stream>>>(t16, nbr_rec, pWb2, bb2, ab2,
                                      h16, y16, y_out, N);
    // p = PReLU(sconv(y))
    k_sconv<<<blkN, thr, 0, stream>>>(y16, nbr_rec, pWp0c, bp0c, ap0,
                                      nullptr, p16, nullptr, N);
    // l8 = p @ Wp0l + bp0l  ([N,512] f16)
    k_lin8<<<blkN, thr, 0, stream>>>(p16, pWp0l, bp0l, l8, N);
    // g = l8[child_pidx, child_bit]
    k_gather<<<dim3((unsigned)(((long)M * 8 + 255) / 256)), 256, 0, stream>>>(
        l8, child_pidx, child_bit, g16, M);
    // q = PReLU(sconv_child(g))
    k_sconv<<<blkM, thr, 0, stream>>>(g16, nbr_child, pWp1c, bp1c, ap1,
                                      nullptr, q16, nullptr, M);
    // logits + CE -> loss
    const float scale = 1.4426950408889634f / (float)BATCH;
    k_logits_ce<<<blkM, thr, 0, stream>>>(q16, pWp1l, bp1l, bins0, oct_kernel,
                                          ref_batch, pnum, M, scale, loss);
}
